// SimpleModelMDLSTM_47614007443741
// MI455X (gfx1250) — matrix operand order repaired and checked by reading, not compiled, not run on hardware
//
#include <hip/hip_runtime.h>
#include <cstdint>
#include <cstddef>
#include <cmath>

// ---------------------------------------------------------------------------
// MDLSTM (2 layers, 4 directions, hidden=11) for MI455X / gfx1250, wave32.
//
// Pipeline:
//   k_reduce  : x (256,1,32,2048) --sum H--> xsum_t (2048,256)           [ws +0,   2 MB]
//   k_lstm0   : 64 chains (4 dir x 16 batch-tiles), 1 wave each.
//               WMMA f32_16x16x32_f16 recurrence, gate-grouped N tiles.
//               Stores h (f16, A-operand layout) -> hbuf                 [ws +2MB, 64 MB]
//   k_lstm1   : same, but gx = (sum_d y_d) @ W1 via a second WMMA.
//               Stores fp32 h per direction -> hbuf2                     [ws +66MB,128 MB]
//   k_combine : out[b][o][w] = sum_d hbuf2[d]                            (deterministic)
//
// Requires ws_size >= ~194 MB.
// ---------------------------------------------------------------------------

#define WD   2048
#define NB   256
#define NOUT 11
#define NG   55

typedef __attribute__((ext_vector_type(16))) _Float16 v16h;
typedef __attribute__((ext_vector_type(8)))  _Float16 v8h;
typedef __attribute__((ext_vector_type(8)))  float    v8f;
typedef __attribute__((ext_vector_type(4)))  float    v4f;

// NOTE: guard on __HIP_DEVICE_COMPILE__ so the host pass never sees the
// amdgcn builtins / device-only intrinsics.
static __device__ __forceinline__ float fexp2(float x) {
#if defined(__HIP_DEVICE_COMPILE__) && __has_builtin(__builtin_amdgcn_exp2f)
  return __builtin_amdgcn_exp2f(x);   // v_exp_f32
#else
  return exp2f(x);
#endif
}
static __device__ __forceinline__ float frcp(float x) {
#if defined(__HIP_DEVICE_COMPILE__) && __has_builtin(__builtin_amdgcn_rcpf)
  return __builtin_amdgcn_rcpf(x);    // v_rcp_f32
#else
  return 1.0f / x;
#endif
}

static __device__ __forceinline__ float fast_sigmoid(float x) {
  // 1 / (1 + 2^(-x*log2 e))
  return frcp(1.0f + fexp2(x * -1.44269504088896340736f));
}
static __device__ __forceinline__ float fast_tanh(float x) {
  // 1 - 2 / (2^(2x*log2 e) + 1); saturates correctly at +/-1
  return 1.0f - 2.0f * frcp(1.0f + fexp2(x * 2.88539008177792681472f));
}

// Gate-grouped column permutation: N-tile t, lane col n -> original gate col.
// tiles: 0=i (0..10), 1=f_w (11..21), 2=o (33..43), 3=a (44..54). f_h (22..32)
// is dropped: with height==1 the vertical cell state is zero.
static __device__ __forceinline__ int gate_col(int t, int n) {
  const int base = (t == 0) ? 0 : ((t == 1) ? 11 : ((t == 2) ? 33 : 44));
  return base + n;
}

// --------------------------------------------------------------------------
// Kernel 1: sum over H (32), transpose to (Wd, B) for broadcast-friendly loads
// --------------------------------------------------------------------------
__global__ __launch_bounds__(256) void k_reduce(const float* __restrict__ x,
                                                float* __restrict__ xsum) {
  const int w = blockIdx.x * 256 + threadIdx.x;
  const int b = blockIdx.y;
  const float* p = x + (size_t)b * 32 * WD + w;
  float s = 0.0f;
#pragma unroll
  for (int h = 0; h < 32; ++h) s += p[(size_t)h * WD];
  xsum[(size_t)w * NB + b] = s;
}

// --------------------------------------------------------------------------
// Kernel 2: layer-0 recurrence. 64 blocks x 32 threads (1 wave per chain).
// chain = d*16 + bt. Dirs 0,2 scan forward; dirs 1,3 backward.
// --------------------------------------------------------------------------
__global__ __launch_bounds__(32) void k_lstm0(const float* __restrict__ xsum,
                                              const float* __restrict__ W0,
                                              const float* __restrict__ U0,
                                              const float* __restrict__ b0,
                                              _Float16* __restrict__ hbuf) {
  const int chain = blockIdx.x;
  const int d  = chain >> 4;
  const int bt = chain & 15;
  const int l  = threadIdx.x & 31;
  const int g  = l >> 4;        // half-wave group
  const int n  = l & 15;        // N column (C/D) == M row (A)
  const bool fwd = ((d & 1) == 0);
  const bool vn  = (n < NOUT);

  __shared__ __align__(16) _Float16 h16[256];   // [m][o] 16x16 f16

  // ---- per-lane constants: permuted W0/b0 scalars + U0 B-operand tiles ----
  float w0c[4], b0c[4];
  v16h  Ub[4];
#pragma unroll
  for (int t = 0; t < 4; ++t) {
    const int col = gate_col(t, n);
    w0c[t] = vn ? W0[d * NG + col] : 0.0f;
    b0c[t] = vn ? b0[d * NG + col] : 0.0f;
    v16h u = {};
#pragma unroll
    for (int e = 0; e < 16; ++e) {
      const int K = (e < 8) ? (8 * g + e) : (16 + 8 * g + (e - 8));   // B layout: table order, K=8g+e (e<8), 16+8g+e-8 (e>=8)
      u[e] = (vn && K < NOUT) ? (_Float16)U0[(d * NOUT + K) * NG + col]
                              : (_Float16)0.0f;
    }
    Ub[t] = u;
  }

  float c[8];
#pragma unroll
  for (int v = 0; v < 8; ++v) c[v] = 0.0f;
  v16h hA = {};                 // h_{-1} = 0 (high K halves stay zero forever)

  const float* xb = xsum + bt * 16 + 8 * g;   // + w*NB -> rows m=8g..8g+7
  const int wprime = fwd ? 0 : (WD - 1);
  v4f xs0 = *(const v4f*)(xb + (size_t)wprime * NB);
  v4f xs1 = *(const v4f*)(xb + (size_t)wprime * NB + 4);

  _Float16* hb = hbuf + (size_t)chain * WD * 256 + l * 8;

  for (int t = 0; t < WD; ++t) {
    const int w  = fwd ? t : (WD - 1 - t);
    const int tn = (t < WD - 1) ? t + 1 : t;
    const int wn = fwd ? tn : (WD - 1 - tn);
    // software pipeline: next step's input in flight while we compute
    v4f xs0n = *(const v4f*)(xb + (size_t)wn * NB);
    v4f xs1n = *(const v4f*)(xb + (size_t)wn * NB + 4);
    if (t + 8 < WD) {
      const int wf = fwd ? (t + 8) : (WD - 9 - t);
      __builtin_prefetch(xb + (size_t)wf * NB, 0, 0);
    }

    // gx tiles (Cin==1): gx = xs*W0p + b0p, then acc = h @ U0p + gx via WMMA
    v8f acc[4];
#pragma unroll
    for (int tt = 0; tt < 4; ++tt) {
      v8f cg;
#pragma unroll
      for (int v = 0; v < 4; ++v) {
        cg[v]     = __builtin_fmaf(xs0[v], w0c[tt], b0c[tt]);
        cg[v + 4] = __builtin_fmaf(xs1[v], w0c[tt], b0c[tt]);
      }
      acc[tt] = __builtin_amdgcn_wmma_f32_16x16x32_f16(
          false, hA, false, Ub[tt], (short)0, cg, false, false);
    }

    // gate math: tiles 0..3 = {i, f_w, o, a} for THIS lane's cell -> no shuffles
    float hv[8];
#pragma unroll
    for (int v = 0; v < 8; ++v) {
      const float ig = fast_sigmoid(acc[0][v]);
      const float fg = fast_sigmoid(acc[1][v]);
      const float og = fast_sigmoid(acc[2][v]);
      const float ag = fast_tanh(acc[3][v]);
      c[v]  = __builtin_fmaf(ig, ag, fg * c[v]);
      hv[v] = og * fast_tanh(c[v]);
    }

    // D-layout -> A-layout via LDS (same-wave LDS is in-order)
#pragma unroll
    for (int v = 0; v < 8; ++v)
      h16[(v + 8 * g) * 16 + n] = (_Float16)hv[v];
    asm volatile("" ::: "memory");
    const v8h lo = *(const v8h*)&h16[n * 16 + 8 * g];   // ds_load_b128
    asm volatile("" ::: "memory");
#pragma unroll
    for (int e = 0; e < 8; ++e) hA[e] = lo[e];

    // store h_t in A-operand f16 layout for layer 1 (coalesced b128)
    *(v8h*)(hb + (size_t)w * 256) = lo;

    xs0 = xs0n; xs1 = xs1n;
  }
}

// --------------------------------------------------------------------------
// Kernel 3: layer-1 recurrence. y = sum_d h0_d (f16 A-operand), gx = y@W1+b1
// via WMMA, then the usual recurrent WMMA. 8 WMMAs/step.
// --------------------------------------------------------------------------
__global__ __launch_bounds__(32) void k_lstm1(const _Float16* __restrict__ hbuf,
                                              const float* __restrict__ W1,
                                              const float* __restrict__ U1,
                                              const float* __restrict__ b1,
                                              float* __restrict__ hbuf2) {
  const int chain = blockIdx.x;
  const int d  = chain >> 4;
  const int bt = chain & 15;
  const int l  = threadIdx.x & 31;
  const int g  = l >> 4;
  const int n  = l & 15;
  const bool fwd = ((d & 1) == 0);
  const bool vn  = (n < NOUT);

  __shared__ __align__(16) _Float16 h16[256];

  float bC[4];
  v16h  Wb[4], Ub[4];
#pragma unroll
  for (int t = 0; t < 4; ++t) {
    const int col = gate_col(t, n);
    bC[t] = vn ? b1[d * NG + col] : 0.0f;
    v16h wv = {}, uv = {};
#pragma unroll
    for (int e = 0; e < 16; ++e) {
      const int K = (e < 8) ? (8 * g + e) : (16 + 8 * g + (e - 8));
      const bool kv = (vn && K < NOUT);
      wv[e] = kv ? (_Float16)W1[(d * NOUT + K) * NG + col] : (_Float16)0.0f;
      uv[e] = kv ? (_Float16)U1[(d * NOUT + K) * NG + col] : (_Float16)0.0f;
    }
    Wb[t] = wv; Ub[t] = uv;
  }

  float c[8];
#pragma unroll
  for (int v = 0; v < 8; ++v) c[v] = 0.0f;
  v16h hA = {};

  const _Float16* yp0 = hbuf + (size_t)(0 * 16 + bt) * WD * 256 + l * 8;
  const _Float16* yp1 = hbuf + (size_t)(1 * 16 + bt) * WD * 256 + l * 8;
  const _Float16* yp2 = hbuf + (size_t)(2 * 16 + bt) * WD * 256 + l * 8;
  const _Float16* yp3 = hbuf + (size_t)(3 * 16 + bt) * WD * 256 + l * 8;

  const int wprime = fwd ? 0 : (WD - 1);
  v8h y0 = *(const v8h*)(yp0 + (size_t)wprime * 256);
  v8h y1 = *(const v8h*)(yp1 + (size_t)wprime * 256);
  v8h y2 = *(const v8h*)(yp2 + (size_t)wprime * 256);
  v8h y3 = *(const v8h*)(yp3 + (size_t)wprime * 256);

  float* ob = hbuf2 + (size_t)chain * WD * 256 + n * 16 + 8 * g;

  for (int t = 0; t < WD; ++t) {
    const int w  = fwd ? t : (WD - 1 - t);
    const int tn = (t < WD - 1) ? t + 1 : t;
    const int wn = fwd ? tn : (WD - 1 - tn);
    v8h y0n = *(const v8h*)(yp0 + (size_t)wn * 256);
    v8h y1n = *(const v8h*)(yp1 + (size_t)wn * 256);
    v8h y2n = *(const v8h*)(yp2 + (size_t)wn * 256);
    v8h y3n = *(const v8h*)(yp3 + (size_t)wn * 256);
    if (t + 8 < WD) {
      const int wf = fwd ? (t + 8) : (WD - 9 - t);
      __builtin_prefetch(yp0 + (size_t)wf * 256, 0, 0);
      __builtin_prefetch(yp1 + (size_t)wf * 256, 0, 0);
      __builtin_prefetch(yp2 + (size_t)wf * 256, 0, 0);
      __builtin_prefetch(yp3 + (size_t)wf * 256, 0, 0);
    }

    // y = sum over the 4 layer-0 directions (packed f16 adds)
    const v8h ys = (y0 + y1) + (y2 + y3);
    v16h yA = {};
#pragma unroll
    for (int e = 0; e < 8; ++e) yA[e] = ys[e];

    v8f acc[4];
#pragma unroll
    for (int tt = 0; tt < 4; ++tt) {
      v8f cb;
#pragma unroll
      for (int v = 0; v < 8; ++v) cb[v] = bC[tt];
      const v8f gx = __builtin_amdgcn_wmma_f32_16x16x32_f16(
          false, yA, false, Wb[tt], (short)0, cb, false, false);
      acc[tt] = __builtin_amdgcn_wmma_f32_16x16x32_f16(
          false, hA, false, Ub[tt], (short)0, gx, false, false);
    }

    float hv[8];
#pragma unroll
    for (int v = 0; v < 8; ++v) {
      const float ig = fast_sigmoid(acc[0][v]);
      const float fg = fast_sigmoid(acc[1][v]);
      const float og = fast_sigmoid(acc[2][v]);
      const float ag = fast_tanh(acc[3][v]);
      c[v]  = __builtin_fmaf(ig, ag, fg * c[v]);
      hv[v] = og * fast_tanh(c[v]);
    }

    // fp32 output h per direction (combined deterministically in k_combine)
    v4f o0, o1;
#pragma unroll
    for (int v = 0; v < 4; ++v) { o0[v] = hv[v]; o1[v] = hv[v + 4]; }
    *(v4f*)(ob + (size_t)w * 256)     = o0;
    *(v4f*)(ob + (size_t)w * 256 + 4) = o1;

    // rebuild A operand for next step
#pragma unroll
    for (int v = 0; v < 8; ++v)
      h16[(v + 8 * g) * 16 + n] = (_Float16)hv[v];
    asm volatile("" ::: "memory");
    const v8h lo = *(const v8h*)&h16[n * 16 + 8 * g];
    asm volatile("" ::: "memory");
#pragma unroll
    for (int e = 0; e < 8; ++e) hA[e] = lo[e];

    y0 = y0n; y1 = y1n; y2 = y2n; y3 = y3n;
  }
}

// --------------------------------------------------------------------------
// Kernel 4: deterministic 4-direction sum -> out (B, OUT, Wd) fp32
// hbuf2 step layout: [n(=o)][m] 16x16 floats. m is the fast thread index ->
// coalesced 64B reads per half-wave.
// --------------------------------------------------------------------------
__global__ __launch_bounds__(256) void k_combine(const float* __restrict__ hbuf2,
                                                 float* __restrict__ out) {
  const int m  = threadIdx.x & 15;
  const int wi = threadIdx.x >> 4;
  const int w  = blockIdx.x * 16 + wi;
  const int o  = blockIdx.y;        // 0..10
  const int bt = blockIdx.z;        // 0..15
  const size_t base = (size_t)w * 256 + (size_t)o * 16 + m;
  float s = 0.0f;
#pragma unroll
  for (int dd = 0; dd < 4; ++dd)
    s += hbuf2[(size_t)(dd * 16 + bt) * WD * 256 + base];
  out[((size_t)(bt * 16 + m) * NOUT + o) * WD + w] = s;
}

// --------------------------------------------------------------------------
extern "C" void kernel_launch(void* const* d_in, const int* in_sizes, int n_in,
                              void* d_out, int out_size, void* d_ws, size_t ws_size,
                              hipStream_t stream) {
  (void)in_sizes; (void)n_in; (void)out_size; (void)ws_size;
  const float* x  = (const float*)d_in[0];
  const float* W0 = (const float*)d_in[1];
  const float* U0 = (const float*)d_in[2];
  const float* b0 = (const float*)d_in[3];
  const float* W1 = (const float*)d_in[4];
  const float* U1 = (const float*)d_in[5];
  const float* b1 = (const float*)d_in[6];
  float* out = (float*)d_out;

  char* ws = (char*)d_ws;
  float*    xsum  = (float*)ws;                                  //   2 MB
  _Float16* hbuf  = (_Float16*)(ws + (size_t)2  * 1024 * 1024);  //  64 MB
  float*    hbuf2 = (float*)   (ws + (size_t)66 * 1024 * 1024);  // 128 MB
  // total scratch requirement: 194 MB

  k_reduce <<<dim3(WD / 256, NB), 256, 0, stream>>>(x, xsum);
  k_lstm0  <<<64, 32, 0, stream>>>(xsum, W0, U0, b0, hbuf);
  k_lstm1  <<<64, 32, 0, stream>>>(hbuf, W1, U1, b1, hbuf2);
  k_combine<<<dim3(WD / 16, NOUT, 16), 256, 0, stream>>>(hbuf2, out);
}
